// Attention_65927747993826
// MI455X (gfx1250) — matrix operand order repaired and checked by reading, not compiled, not run on hardware
//
#include <hip/hip_runtime.h>
#include <hip/hip_bf16.h>

typedef __bf16 bf16_t;
typedef __attribute__((ext_vector_type(16))) __bf16 v16bf;
typedef __attribute__((ext_vector_type(8)))  __bf16 v8bf;
typedef __attribute__((ext_vector_type(8)))  float  v8f;

#define EMBED   2048
#define NHEADS  32
#define NGROUPS 8
#define HEADSZ  64
#define KVDIM   (NGROUPS * HEADSZ)   // 512
#define BATCH   2
#define SEQ     2048
#define MTOT    (BATCH * SEQ)        // 4096
#define QK_SCALE 0.125f              // 1/sqrt(64)

union AB16 { v16bf v; v8bf h[2]; };

static __device__ __forceinline__ v8f vzero8() {
    v8f z;
#pragma unroll
    for (int i = 0; i < 8; ++i) z[i] = 0.0f;
    return z;
}

// Generic pointer to a __shared__ object: LDS byte offset lives in bits [31:0].
static __device__ __forceinline__ unsigned lds_off(const void* p) {
    return (unsigned)(size_t)p;
}

// CDNA5 async copy: 16B global -> LDS, tracked by ASYNCcnt (no VGPR round-trip).
static __device__ __forceinline__ void async_copy_b128(const void* gaddr, unsigned lds) {
    asm volatile("global_load_async_to_lds_b128 %0, %1, off"
                 :: "v"(lds), "v"(gaddr) : "memory");
}
static __device__ __forceinline__ void wait_asynccnt0() {
    asm volatile("s_wait_asynccnt 0" ::: "memory");
}

// ---------------------------------------------------------------------------
// fp32 -> bf16 conversion (grid-stride)
// ---------------------------------------------------------------------------
__global__ void cvt_f32_to_bf16(const float* __restrict__ in,
                                bf16_t* __restrict__ out, int n) {
    int i = blockIdx.x * blockDim.x + threadIdx.x;
    int stride = gridDim.x * blockDim.x;
    for (; i < n; i += stride) out[i] = (bf16_t)in[i];
}

// ---------------------------------------------------------------------------
// Tiled bf16 GEMM: C[M,N] = A[M,K] @ B[K,N], A/B bf16 row-major.
// Block = 256 threads (8 waves), tile 128x128, K-step 32.
// Wave (wm,wn) computes a 32x64 sub-tile: 2 A-frags x 4 B-frags -> 8 WMMA/iter.
// A staged via global_load_async_to_lds_b128; B staged transposed (register path).
// ---------------------------------------------------------------------------
template <bool STORE_BF16>
__global__ __launch_bounds__(256) void gemm_bf16(
    const bf16_t* __restrict__ A, const bf16_t* __restrict__ B,
    int M, int N, int K, const float* __restrict__ bias,
    float* __restrict__ Cf, bf16_t* __restrict__ Cb) {
    __shared__ alignas(32) bf16_t As[128 * 32];  // row-major [m][k]
    __shared__ alignas(32) bf16_t Bt[128 * 32];  // transposed [n][k]

    const int tid   = threadIdx.x;
    const int wave  = tid >> 5;        // 0..7
    const int lane  = tid & 31;
    const int lhalf = lane >> 4;       // 0: lanes 0-15, 1: lanes 16-31
    const int l16   = lane & 15;
    const int wm    = wave & 3;        // M strip of 32 rows
    const int wn    = wave >> 2;       // N strip of 64 cols
    const int m0    = blockIdx.y * 128;
    const int n0    = blockIdx.x * 128;

    v8f acc[2][4];
#pragma unroll
    for (int mt = 0; mt < 2; ++mt)
#pragma unroll
        for (int nt = 0; nt < 4; ++nt) acc[mt][nt] = vzero8();

    // staging assignments (constant per thread)
    const int arow  = tid >> 1;            // 0..127
    const int aoff  = (tid & 1) * 16;      // 0 or 16 (halfs)
    const unsigned lds_a = lds_off(&As[arow * 32 + aoff]);
    const int bkrow = tid >> 3;            // 0..31
    const int bnoff = (tid & 7) * 16;      // 0..112 (halfs)

    for (int k0 = 0; k0 < K; k0 += 32) {
        // ---- A tile 128x32: async global->LDS (2 x b128 per thread) ----
        const bf16_t* asrc = A + (size_t)(m0 + arow) * K + k0 + aoff;
        async_copy_b128(asrc, lds_a);
        async_copy_b128(asrc + 8, lds_a + 16);
        // ---- B tile 32x128 -> Bt[n][k] (needs transpose: register path) ----
        const bf16_t* bsrc = B + (size_t)(k0 + bkrow) * N + n0 + bnoff;
        v8bf b0 = *(const v8bf*)bsrc;
        v8bf b1 = *(const v8bf*)(bsrc + 8);
        if (k0 + 32 < K) {  // prefetch next K-step lines into GL2
            __builtin_prefetch(asrc + K ? asrc + 32 : asrc, 0, 3);
            __builtin_prefetch(bsrc + (size_t)32 * N, 0, 3);
        }
#pragma unroll
        for (int i = 0; i < 8; ++i) {
            Bt[(bnoff + i)     * 32 + bkrow] = b0[i];
            Bt[(bnoff + 8 + i) * 32 + bkrow] = b1[i];
        }
        wait_asynccnt0();
        __syncthreads();

        // ---- A fragments (contiguous k order shared with Bt: K chunks {0-15}/{16-31}) ----
        AB16 af[2];
#pragma unroll
        for (int mt = 0; mt < 2; ++mt) {
            const int r = wm * 32 + mt * 16 + l16;
            af[mt].h[0] = *(const v8bf*)&As[r * 32 + (lhalf ? 16 : 0)];
            af[mt].h[1] = *(const v8bf*)&As[r * 32 + (lhalf ? 24 : 8)];
        }
#pragma unroll
        for (int nt = 0; nt < 4; ++nt) {
            const v16bf bf =
                *(const v16bf*)&Bt[(wn * 64 + nt * 16 + l16) * 32 + (lhalf ? 16 : 0)];
#pragma unroll
            for (int mt = 0; mt < 2; ++mt)
                acc[mt][nt] = __builtin_amdgcn_wmma_f32_16x16x32_bf16(
                    false, af[mt].v, false, bf, (short)0, acc[mt][nt], false, false);
        }
        __syncthreads();
    }

    // ---- epilogue: D layout row M = r + 8*lhalf, col N = l16 ----
#pragma unroll
    for (int mt = 0; mt < 2; ++mt) {
#pragma unroll
        for (int nt = 0; nt < 4; ++nt) {
            const int n = n0 + wn * 64 + nt * 16 + l16;
            const float bv = bias ? bias[n] : 0.0f;
#pragma unroll
            for (int r = 0; r < 8; ++r) {
                const int m = m0 + wm * 32 + mt * 16 + r + 8 * lhalf;
                const float val = acc[mt][nt][r] + bv;
                if constexpr (STORE_BF16) Cb[(size_t)m * N + n] = (bf16_t)val;
                else                      Cf[(size_t)m * N + n] = val;
            }
        }
    }
}

// ---------------------------------------------------------------------------
// Flash-attention (causal, GQA). Block = 128 threads = 4 waves handling
// 64 consecutive queries of one (batch, head); key blocks of 32 staged in LDS.
// K tile staged with async global->LDS; V tile transposed via registers.
// q layout: [B,S,H,D] bf16   k/v layout: [B,S,G,D] bf16   o: [B,S,H,D] bf16
// ---------------------------------------------------------------------------
__global__ __launch_bounds__(128) void attn_flash(
    const bf16_t* __restrict__ q, const bf16_t* __restrict__ k,
    const bf16_t* __restrict__ v, bf16_t* __restrict__ o) {
    __shared__ alignas(32) bf16_t Ks[32 * 64];     // [key][d]
    __shared__ alignas(32) bf16_t Vt[64 * 32];     // [d][key] (transposed)
    __shared__ alignas(32) bf16_t Ps[4][16 * 32];  // per-wave P tile [q][key]

    const int tid   = threadIdx.x;
    const int wave  = tid >> 5;
    const int lane  = tid & 31;
    const int lhalf = lane >> 4;
    const int l16   = lane & 15;

    const int b    = blockIdx.z;
    const int h    = blockIdx.y;
    const int g    = h / (NHEADS / NGROUPS);
    const int qblk = blockIdx.x;              // 64 queries per block
    const int q0   = qblk * 64 + wave * 16;   // this wave's query tile base

    const bf16_t* qbase = q + ((size_t)(b * SEQ) * NHEADS  + h) * HEADSZ;
    const bf16_t* kbase = k + ((size_t)(b * SEQ) * NGROUPS + g) * HEADSZ;
    const bf16_t* vbase = v + ((size_t)(b * SEQ) * NGROUPS + g) * HEADSZ;

    // ---- Q fragments (A layout), row = q0 + l16, d = 0..63 ----
    AB16 qf0, qf1;
    {
        const bf16_t* qr = qbase + (size_t)(q0 + l16) * (NHEADS * HEADSZ);
        qf0.h[0] = *(const v8bf*)(qr + (lhalf ? 16 : 0));
        qf0.h[1] = *(const v8bf*)(qr + (lhalf ? 24 : 8));
        qf1.h[0] = *(const v8bf*)(qr + 32 + (lhalf ? 16 : 0));
        qf1.h[1] = *(const v8bf*)(qr + 32 + (lhalf ? 24 : 8));
    }

    v8f oacc[4];
#pragma unroll
    for (int t = 0; t < 4; ++t) oacc[t] = vzero8();
    float m_run[8], l_run[8];
#pragma unroll
    for (int r = 0; r < 8; ++r) { m_run[r] = -1e30f; l_run[r] = 0.0f; }

    // staging assignments
    const int srow = tid >> 2;          // 0..31 key rows
    const int soff = (tid & 3) * 16;    // 0,16,32,48 d offset (halfs)
    const unsigned lds_k = lds_off(&Ks[srow * 64 + soff]);

    const int nkb = 2 * qblk + 2;  // causal: key blocks for queries < (qblk+1)*64
    for (int kb = 0; kb < nkb; ++kb) {
        const int kk0 = kb * 32;
        // ---- K tile 32x64: async global->LDS ----
        const bf16_t* ks = kbase + (size_t)(kk0 + srow) * KVDIM + soff;
        async_copy_b128(ks, lds_k);
        async_copy_b128(ks + 8, lds_k + 16);
        // ---- V tile 32x64 -> Vt[d][key] (transpose via registers) ----
        const bf16_t* vs = vbase + (size_t)(kk0 + srow) * KVDIM + soff;
        v8bf va = *(const v8bf*)vs;
        v8bf vb = *(const v8bf*)(vs + 8);
#pragma unroll
        for (int i = 0; i < 8; ++i) {
            Vt[(soff + i)     * 32 + srow] = va[i];
            Vt[(soff + 8 + i) * 32 + srow] = vb[i];
        }
        wait_asynccnt0();
        __syncthreads();

        if (kk0 <= q0 + 15) {  // this wave has unmasked keys in the block
            // ---- scores S = Q @ K^T  (2 key n-tiles x 2 d k-chunks) ----
            v8f s[2];
#pragma unroll
            for (int nt = 0; nt < 2; ++nt) {
                s[nt] = vzero8();
                const v16bf kf0 = *(const v16bf*)&Ks[(nt * 16 + l16) * 64 + (lhalf ? 16 : 0)];
                const v16bf kf1 = *(const v16bf*)&Ks[(nt * 16 + l16) * 64 + 32 + (lhalf ? 16 : 0)];
                s[nt] = __builtin_amdgcn_wmma_f32_16x16x32_bf16(
                    false, qf0.v, false, kf0, (short)0, s[nt], false, false);
                s[nt] = __builtin_amdgcn_wmma_f32_16x16x32_bf16(
                    false, qf1.v, false, kf1, (short)0, s[nt], false, false);
            }
            // ---- scale + causal mask + online softmax ----
#pragma unroll
            for (int r = 0; r < 8; ++r) {
                const int qrow = q0 + r + 8 * lhalf;
#pragma unroll
                for (int nt = 0; nt < 2; ++nt) {
                    const int key = kk0 + nt * 16 + l16;
                    s[nt][r] = (key <= qrow) ? s[nt][r] * QK_SCALE : -1e30f;
                }
                float mloc = fmaxf(s[0][r], s[1][r]);
#pragma unroll
                for (int off = 1; off < 16; off <<= 1)
                    mloc = fmaxf(mloc, __shfl_xor(mloc, off, 32));
                const float mnew = fmaxf(m_run[r], mloc);
                const float corr = __expf(m_run[r] - mnew);
                const float p0 = __expf(s[0][r] - mnew);
                const float p1 = __expf(s[1][r] - mnew);
                float rs = p0 + p1;
#pragma unroll
                for (int off = 1; off < 16; off <<= 1)
                    rs += __shfl_xor(rs, off, 32);
                l_run[r] = l_run[r] * corr + rs;
                m_run[r] = mnew;
#pragma unroll
                for (int t = 0; t < 4; ++t) oacc[t][r] *= corr;
                // scatter P to LDS in [q][key] order (per-wave buffer)
                Ps[wave][(r + 8 * lhalf) * 32 + l16]      = (bf16_t)p0;
                Ps[wave][(r + 8 * lhalf) * 32 + 16 + l16] = (bf16_t)p1;
            }
            // same-wave LDS RAW: DS ops are in-order; compiler inserts s_wait_dscnt
            AB16 pf;
            pf.h[0] = *(const v8bf*)&Ps[wave][l16 * 32 + (lhalf ? 16 : 0)];
            pf.h[1] = *(const v8bf*)&Ps[wave][l16 * 32 + (lhalf ? 24 : 8)];
            // ---- O += P @ V ----
#pragma unroll
            for (int t = 0; t < 4; ++t) {
                const v16bf vf = *(const v16bf*)&Vt[(t * 16 + l16) * 32 + (lhalf ? 16 : 0)];
                oacc[t] = __builtin_amdgcn_wmma_f32_16x16x32_bf16(
                    false, pf.v, false, vf, (short)0, oacc[t], false, false);
            }
        }
        __syncthreads();  // unconditional: all waves, before restaging LDS
    }

    // ---- epilogue: normalize and store O as bf16 [B,S,H,D] ----
#pragma unroll
    for (int t = 0; t < 4; ++t) {
        const int d = t * 16 + l16;
#pragma unroll
        for (int r = 0; r < 8; ++r) {
            const int qrow = q0 + r + 8 * lhalf;
            const float ov = oacc[t][r] / l_run[r];
            o[((size_t)(b * SEQ + qrow) * NHEADS + h) * HEADSZ + d] = (bf16_t)ov;
        }
    }
}

// ---------------------------------------------------------------------------
extern "C" void kernel_launch(void* const* d_in, const int* in_sizes, int n_in,
                              void* d_out, int out_size, void* d_ws, size_t ws_size,
                              hipStream_t stream) {
    const float* x  = (const float*)d_in[0];
    const float* Wq = (const float*)d_in[1];
    const float* Wk = (const float*)d_in[2];
    const float* Wv = (const float*)d_in[3];
    const float* Wo = (const float*)d_in[4];
    const float* bo = (const float*)d_in[5];
    // d_in[6] = causal mask: statically known, ignored.
    float* out = (float*)d_out;

    char* ws = (char*)d_ws;
    size_t off = 0;
    bf16_t* xb  = (bf16_t*)(ws + off); off += (size_t)MTOT * EMBED * 2;
    bf16_t* Wqb = (bf16_t*)(ws + off); off += (size_t)EMBED * EMBED * 2;
    bf16_t* Wkb = (bf16_t*)(ws + off); off += (size_t)EMBED * KVDIM * 2;
    bf16_t* Wvb = (bf16_t*)(ws + off); off += (size_t)EMBED * KVDIM * 2;
    bf16_t* Wob = (bf16_t*)(ws + off); off += (size_t)EMBED * EMBED * 2;
    bf16_t* qb  = (bf16_t*)(ws + off); off += (size_t)MTOT * EMBED * 2;
    bf16_t* kbuf= (bf16_t*)(ws + off); off += (size_t)MTOT * KVDIM * 2;
    bf16_t* vbuf= (bf16_t*)(ws + off); off += (size_t)MTOT * KVDIM * 2;
    bf16_t* ob  = (bf16_t*)(ws + off); off += (size_t)MTOT * EMBED * 2;

    auto cvt = [&](const float* src, bf16_t* dst, int n) {
        int blocks = (n + 255) / 256;
        if (blocks > 4096) blocks = 4096;
        cvt_f32_to_bf16<<<blocks, 256, 0, stream>>>(src, dst, n);
    };
    cvt(x,  xb,  MTOT * EMBED);
    cvt(Wq, Wqb, EMBED * EMBED);
    cvt(Wk, Wkb, EMBED * KVDIM);
    cvt(Wv, Wvb, EMBED * KVDIM);
    cvt(Wo, Wob, EMBED * EMBED);

    // q = x @ Wq     [4096 x 2048] x [2048 x 2048]
    {
        dim3 grid(EMBED / 128, MTOT / 128);
        gemm_bf16<true><<<grid, 256, 0, stream>>>(xb, Wqb, MTOT, EMBED, EMBED,
                                                  nullptr, nullptr, qb);
    }
    // k, v = x @ Wk, x @ Wv   [4096 x 2048] x [2048 x 512]
    {
        dim3 grid(KVDIM / 128, MTOT / 128);
        gemm_bf16<true><<<grid, 256, 0, stream>>>(xb, Wkb, MTOT, KVDIM, EMBED,
                                                  nullptr, nullptr, kbuf);
        gemm_bf16<true><<<grid, 256, 0, stream>>>(xb, Wvb, MTOT, KVDIM, EMBED,
                                                  nullptr, nullptr, vbuf);
    }
    // causal GQA flash attention
    {
        dim3 grid(SEQ / 64, NHEADS, BATCH);
        attn_flash<<<grid, 128, 0, stream>>>(qb, kbuf, vbuf, ob);
    }
    // out = attn @ Wo + bo   (fp32 result)
    {
        dim3 grid(EMBED / 128, MTOT / 128);
        gemm_bf16<false><<<grid, 256, 0, stream>>>(ob, Wob, MTOT, EMBED, EMBED,
                                                   bo, out, nullptr);
    }
}
